// MPNN_70342974374417
// MI455X (gfx1250) — matrix operand order repaired and checked by reading, not compiled, not run on hardware
//
#include <hip/hip_runtime.h>

typedef __bf16 bf16;
typedef __attribute__((ext_vector_type(16))) __bf16 bf16x16;
typedef __attribute__((ext_vector_type(8)))  float  floatx8;

constexpr int kN = 32768;   // nodes
constexpr int kE = 65536;   // edges
constexpr int kG = 1024;    // graphs
constexpr int kP = 32;      // nodes per graph
constexpr int kATOM = 29;
constexpr int kD = 64;
constexpr int kDENSE = 512;
constexpr int kTAB = 7;
constexpr int kCLSK = 96;   // 64 pooled + 7 tab, zero-padded (K multiple of 32)

// ---------------------------------------------------------------------------
// WMMA helpers (CDNA5 16x16x32 bf16, wave32; A and B share contiguous k = 16*(lane>>4) + j)
// ---------------------------------------------------------------------------
static __device__ inline floatx8 wmma_bf16(bf16x16 a, bf16x16 b, floatx8 c) {
  return __builtin_amdgcn_wmma_f32_16x16x32_bf16(false, a, false, b,
                                                 (short)0, c, false, false);
}

// A (16x32 bf16) from fp32 activations: lane L -> row = L&15,
// K(j) = kk + 16*(L>>4) + j  (four contiguous float4)
static __device__ inline bf16x16 load_A_frag(const float* __restrict__ A,
                                             int row0, int K, int kk, int lane) {
  const int hi = lane >> 4;
  const float* a = A + (unsigned)(row0 + (lane & 15)) * (unsigned)K + kk + hi * 16;
  const float4 p0 = *(const float4*)(a);
  const float4 p1 = *(const float4*)(a + 4);
  const float4 p2 = *(const float4*)(a + 8);
  const float4 p3 = *(const float4*)(a + 12);
  bf16x16 f;
  f[0]  = (bf16)p0.x; f[1]  = (bf16)p0.y; f[2]  = (bf16)p0.z; f[3]  = (bf16)p0.w;
  f[4]  = (bf16)p1.x; f[5]  = (bf16)p1.y; f[6]  = (bf16)p1.z; f[7]  = (bf16)p1.w;
  f[8]  = (bf16)p2.x; f[9]  = (bf16)p2.y; f[10] = (bf16)p2.z; f[11] = (bf16)p2.w;
  f[12] = (bf16)p3.x; f[13] = (bf16)p3.y; f[14] = (bf16)p3.z; f[15] = (bf16)p3.w;
  return f;
}

// B fragments are pre-swizzled by pack_B into fragment-major order:
// Bp[((kc*nCT + ct)*32 + lane)*16 + j]  ->  one contiguous 32B load per lane.
static __device__ inline bf16x16 load_Bp(const bf16* __restrict__ Bp,
                                         int kc, int ct, int nCT, int lane) {
  return *(const bf16x16*)(Bp + (((unsigned)(kc * nCT + ct) * 32u + (unsigned)lane) << 4));
}

// ---------------------------------------------------------------------------
// GEMM: C[M,Ncols] = act(A[M,K](f32) @ B[K,Ncols](bf16 packed) + bias)
// grid = (M/64, Ncols/64), block = (32,4): each wave owns 16 rows x 64 cols.
// ---------------------------------------------------------------------------
__global__ __launch_bounds__(128)
void gemm_bf16(const float* __restrict__ A, const bf16* __restrict__ Bp,
               const float* __restrict__ bias, float* __restrict__ C,
               int Ncols, int K, int relu) {
  const int lane = threadIdx.x;
  const int row0 = blockIdx.x * 64 + threadIdx.y * 16;
  const int ct0  = blockIdx.y * 4;     // 16-col tiles
  const int nCT  = Ncols >> 4;
  floatx8 acc[4];
#pragma unroll
  for (int t = 0; t < 4; ++t) acc[t] = floatx8{0.f,0.f,0.f,0.f,0.f,0.f,0.f,0.f};
  const int nKC = K >> 5;
  for (int kc = 0; kc < nKC; ++kc) {
    bf16x16 a = load_A_frag(A, row0, K, kc << 5, lane);
#pragma unroll
    for (int t = 0; t < 4; ++t)
      acc[t] = wmma_bf16(a, load_Bp(Bp, kc, ct0 + t, nCT, lane), acc[t]);
  }
  const int hi = lane >> 4;
#pragma unroll
  for (int t = 0; t < 4; ++t) {
    const int col = ((ct0 + t) << 4) + (lane & 15);
    const float bs = bias ? bias[col] : 0.f;
    float* cp = C + (unsigned)(row0 + hi * 8) * (unsigned)Ncols + col;
#pragma unroll
    for (int g = 0; g < 8; ++g) {
      float v = acc[t][g] + bs;
      if (relu) v = fmaxf(v, 0.f);
      *cp = v;
      cp += Ncols;
    }
  }
}

// ---------------------------------------------------------------------------
// Fused GRU cell: gi = m@Wih+bih, gh = h@Whh+bhh, gates, write h_next.
// grid = (N/16, 4); block = 32 (one wave). Wih/Whh packed (nCT = 12).
// ---------------------------------------------------------------------------
__global__ __launch_bounds__(32)
void gru_fused(const float* __restrict__ m, const float* __restrict__ h_cur,
               float* __restrict__ h_next,
               const bf16* __restrict__ Wih, const bf16* __restrict__ Whh,
               const float* __restrict__ bih, const float* __restrict__ bhh) {
  const int lane = threadIdx.x;
  const int row0 = blockIdx.x * 16;
  const int f0   = blockIdx.y * 16;

  bf16x16 am0 = load_A_frag(m, row0, kD, 0, lane);
  bf16x16 am1 = load_A_frag(m, row0, kD, 32, lane);
  bf16x16 ah0 = load_A_frag(h_cur, row0, kD, 0, lane);
  bf16x16 ah1 = load_A_frag(h_cur, row0, kD, 32, lane);

  const floatx8 z8 = {0.f,0.f,0.f,0.f,0.f,0.f,0.f,0.f};
  floatx8 gi[3], gh[3];
#pragma unroll
  for (int c = 0; c < 3; ++c) {
    const int ct = c * 4 + (f0 >> 4);
    floatx8 ai = wmma_bf16(am0, load_Bp(Wih, 0, ct, 12, lane), z8);
    ai         = wmma_bf16(am1, load_Bp(Wih, 1, ct, 12, lane), ai);
    gi[c] = ai;
    floatx8 ah = wmma_bf16(ah0, load_Bp(Whh, 0, ct, 12, lane), z8);
    ah         = wmma_bf16(ah1, load_Bp(Whh, 1, ct, 12, lane), ah);
    gh[c] = ah;
  }

  const int hi = lane >> 4;
  const int f  = f0 + (lane & 15);
#pragma unroll
  for (int g = 0; g < 8; ++g) {
    const unsigned row = row0 + g + hi * 8;
    float ir  = gi[0][g] + bih[f];
    float iz  = gi[1][g] + bih[kD + f];
    float in_ = gi[2][g] + bih[2 * kD + f];
    float hr  = gh[0][g] + bhh[f];
    float hz  = gh[1][g] + bhh[kD + f];
    float hn  = gh[2][g] + bhh[2 * kD + f];
    float r  = 1.f / (1.f + __expf(-(ir + hr)));
    float z  = 1.f / (1.f + __expf(-(iz + hz)));
    float nn = tanhf(in_ + r * hn);
    float ho = h_cur[row * kD + f];
    h_next[row * kD + f] = (1.f - z) * nn + z * ho;
  }
}

// ---------------------------------------------------------------------------
// Elementwise / scatter kernels
// ---------------------------------------------------------------------------
__global__ void pad_h0(const float* __restrict__ x, float* __restrict__ h) {
  int idx = blockIdx.x * blockDim.x + threadIdx.x;
  if (idx >= kN * kD) return;
  int n = idx >> 6, f = idx & 63;
  h[idx] = (f < kATOM) ? x[n * kATOM + f] : 0.f;
}

__global__ void m_init(float* __restrict__ m, const float* __restrict__ cb) {
  int idx = blockIdx.x * blockDim.x + threadIdx.x;
  if (idx >= kN * 16) return;
  ((float4*)m)[idx] = ((const float4*)cb)[idx & 15];
}

// msg[e,f] = Y[src,448+f] + sum_k ea[e,k]*Y[src,64k+f]; scatter-add to m[dst,f]
__global__ void edge_combine(const float* __restrict__ Y,
                             const float* __restrict__ ea,
                             const int* __restrict__ ei,
                             float* __restrict__ m) {
  int idx = blockIdx.x * blockDim.x + threadIdx.x;
  if (idx >= kE * 16) return;
  const int e = idx >> 4, q = (idx & 15) << 2;
  const int s = ei[e], d = ei[kE + e];
  const float* y = Y + (unsigned)s * 512u;
  float4 acc = *(const float4*)(y + 448 + q);
  float ek[7];
#pragma unroll
  for (int k = 0; k < 7; ++k) ek[k] = ea[e * 7 + k];
#pragma unroll
  for (int k = 0; k < 7; ++k) {
    const float4 yv = *(const float4*)(y + k * 64 + q);
    acc.x += ek[k] * yv.x; acc.y += ek[k] * yv.y;
    acc.z += ek[k] * yv.z; acc.w += ek[k] * yv.w;
  }
  float* mp = m + (unsigned)d * 64u + q;
  atomicAdd(mp + 0, acc.x);
  atomicAdd(mp + 1, acc.y);
  atomicAdd(mp + 2, acc.z);
  atomicAdd(mp + 3, acc.w);
}

// per-graph per-head attention core (K=8 too small for 16x16x32 WMMA tiles)
__global__ __launch_bounds__(256)
void attn_core(const float* __restrict__ qkv, float* __restrict__ o) {
  const int g = blockIdx.x, p = threadIdx.x, hd = threadIdx.y;
  const float* base = qkv + (unsigned)g * (kP * 192u);
  const float4 q0 = *(const float4*)(base + p * 192 + hd * 8);
  const float4 q1 = *(const float4*)(base + p * 192 + hd * 8 + 4);
  float s[kP], mx = -3.0e38f;
  for (int kk = 0; kk < kP; ++kk) {
    const float4 k0 = *(const float4*)(base + kk * 192 + 64 + hd * 8);
    const float4 k1 = *(const float4*)(base + kk * 192 + 64 + hd * 8 + 4);
    float acc = q0.x*k0.x + q0.y*k0.y + q0.z*k0.z + q0.w*k0.w
              + q1.x*k1.x + q1.y*k1.y + q1.z*k1.z + q1.w*k1.w;
    acc *= 0.35355339059f;   // 1/sqrt(8)
    s[kk] = acc;
    mx = acc > mx ? acc : mx;
  }
  float sum = 0.f;
  for (int kk = 0; kk < kP; ++kk) { s[kk] = __expf(s[kk] - mx); sum += s[kk]; }
  const float inv = 1.f / sum;
  float4 o0 = {0,0,0,0}, o1 = {0,0,0,0};
  for (int kk = 0; kk < kP; ++kk) {
    const float w = s[kk];
    const float4 v0 = *(const float4*)(base + kk * 192 + 128 + hd * 8);
    const float4 v1 = *(const float4*)(base + kk * 192 + 128 + hd * 8 + 4);
    o0.x += w*v0.x; o0.y += w*v0.y; o0.z += w*v0.z; o0.w += w*v0.w;
    o1.x += w*v1.x; o1.y += w*v1.y; o1.z += w*v1.z; o1.w += w*v1.w;
  }
  o0.x *= inv; o0.y *= inv; o0.z *= inv; o0.w *= inv;
  o1.x *= inv; o1.y *= inv; o1.z *= inv; o1.w *= inv;
  float* op = o + ((unsigned)g * kP + p) * 64u + hd * 8;
  *(float4*)(op)     = o0;
  *(float4*)(op + 4) = o1;
}

// one wave per row: residual + layernorm, shuffle reductions over 32 lanes
__global__ __launch_bounds__(256)
void layernorm_res(const float* __restrict__ a, const float* __restrict__ b,
                   const float* __restrict__ gam, const float* __restrict__ bet,
                   float* __restrict__ out) {
  const int row  = blockIdx.x * 8 + threadIdx.y;
  const int lane = threadIdx.x;
  const unsigned base = (unsigned)row * kD;
  float v0 = a[base + lane]      + b[base + lane];
  float v1 = a[base + 32 + lane] + b[base + 32 + lane];
  float s = v0 + v1;
#pragma unroll
  for (int off = 16; off; off >>= 1) s += __shfl_xor(s, off, 32);
  const float mu = s * (1.f / kD);
  const float d0 = v0 - mu, d1 = v1 - mu;
  float vs = d0 * d0 + d1 * d1;
#pragma unroll
  for (int off = 16; off; off >>= 1) vs += __shfl_xor(vs, off, 32);
  const float rs = rsqrtf(vs * (1.f / kD) + 1e-5f);
  out[base + lane]      = d0 * rs * gam[lane]      + bet[lane];
  out[base + 32 + lane] = d1 * rs * gam[32 + lane] + bet[32 + lane];
}

// pooled mean over P, concat tab features, zero-pad K to 96
__global__ void pool_concat(const float* __restrict__ pout, const float* __restrict__ u,
                            float* __restrict__ comb) {
  int idx = blockIdx.x * blockDim.x + threadIdx.x;
  if (idx >= kG * kCLSK) return;
  int g = idx / kCLSK, f = idx % kCLSK;
  float v = 0.f;
  if (f < kD) {
    float s = 0.f;
    for (int p = 0; p < kP; ++p) s += pout[((unsigned)g * kP + p) * 64u + f];
    v = s * (1.f / kP);
  } else if (f < kD + kTAB) {
    v = u[g * kTAB + (f - kD)];
  }
  comb[idx] = v;
}

__global__ void final_dot(const float* __restrict__ c1, const float* __restrict__ W2,
                          const float* __restrict__ b2, float* __restrict__ out) {
  int g = blockIdx.x * blockDim.x + threadIdx.x;
  if (g >= kG) return;
  const float4* cv = (const float4*)(c1 + (unsigned)g * kDENSE);
  const float4* wv = (const float4*)W2;
  float acc = b2[0];
  for (int i = 0; i < kDENSE / 4; ++i) {
    const float4 c = cv[i], w = wv[i];
    acc += c.x*w.x + c.y*w.y + c.z*w.z + c.w*w.w;
  }
  out[g] = acc;
}

// ---------------------------------------------------------------------------
// Weight repacks into WMMA-B fragment order (bf16), zero-padding K.
// out[((kc*nCT + ct)*32 + lane)*16 + j] = W[(kc*32 + (lane>>4)*16 + j), ct*16 + (lane&15)]
// ---------------------------------------------------------------------------
__global__ void pack_B(const float* __restrict__ W, bf16* __restrict__ out,
                       int Ncols, int Kdst, int Ksrc) {
  int idx = blockIdx.x * blockDim.x + threadIdx.x;
  if (idx >= Kdst * Ncols) return;
  const int j    = idx & 15;
  const int lane = (idx >> 4) & 31;
  const int rest = idx >> 9;
  const int nCT  = Ncols >> 4;
  const int ct   = rest % nCT;
  const int kc   = rest / nCT;
  const int k    = kc * 32 + (lane >> 4) * 16 + j;
  const int col  = ct * 16 + (lane & 15);
  out[idx] = (bf16)((k < Ksrc) ? W[k * Ncols + col] : 0.f);
}

// Wcat[d, 64*blk + f] = W_e[blk, 64*d + f] (blk<7) | b_e[64*d + f] (blk==7)
__global__ void pack_wcat_B(const float* __restrict__ We, const float* __restrict__ be,
                            bf16* __restrict__ out) {
  int idx = blockIdx.x * blockDim.x + threadIdx.x;
  if (idx >= kD * 512) return;
  const int j    = idx & 15;
  const int lane = (idx >> 4) & 31;
  const int rest = idx >> 9;
  const int nCT  = 32;                  // 512/16
  const int ct   = rest % nCT;
  const int kc   = rest / nCT;
  const int k    = kc * 32 + (lane >> 4) * 16 + j;   // d dimension (0..63)
  const int col  = ct * 16 + (lane & 15);            // 0..511
  const int blk = col >> 6, f = col & 63;
  const float v = (blk < 7) ? We[blk * (kD * kD) + k * kD + f] : be[k * kD + f];
  out[idx] = (bf16)v;
}

// ---------------------------------------------------------------------------
extern "C" void kernel_launch(void* const* d_in, const int* in_sizes, int n_in,
                              void* d_out, int out_size, void* d_ws, size_t ws_size,
                              hipStream_t stream) {
  (void)in_sizes; (void)n_in; (void)out_size; (void)ws_size;
  const float* x          = (const float*)d_in[0];
  const int*   edge_index = (const int*)  d_in[1];
  const float* edge_attr  = (const float*)d_in[2];
  // d_in[3] = batch (contiguous equal graphs -> unused)
  const float* u          = (const float*)d_in[4];
  const float* W_e        = (const float*)d_in[5];
  const float* b_e        = (const float*)d_in[6];
  const float* conv_bias  = (const float*)d_in[7];
  const float* gru_Wih    = (const float*)d_in[8];
  const float* gru_Whh    = (const float*)d_in[9];
  const float* gru_bih    = (const float*)d_in[10];
  const float* gru_bhh    = (const float*)d_in[11];
  const float* attn_in_w  = (const float*)d_in[12];
  const float* attn_in_b  = (const float*)d_in[13];
  const float* attn_out_w = (const float*)d_in[14];
  const float* attn_out_b = (const float*)d_in[15];
  const float* dp_W1      = (const float*)d_in[16];
  const float* dp_b1      = (const float*)d_in[17];
  const float* dp_W2      = (const float*)d_in[18];
  const float* dp_b2      = (const float*)d_in[19];
  const float* ln1_g      = (const float*)d_in[20];
  const float* ln1_b      = (const float*)d_in[21];
  const float* ln2_g      = (const float*)d_in[22];
  const float* ln2_b      = (const float*)d_in[23];
  const float* cls_W1     = (const float*)d_in[24];
  const float* cls_b1     = (const float*)d_in[25];
  const float* cls_W2     = (const float*)d_in[26];
  const float* cls_b2     = (const float*)d_in[27];
  float* out = (float*)d_out;

  // workspace carve-up (~146 MB)
  char* base = (char*)d_ws;
  size_t off = 0;
  auto carve = [&](size_t bytes) {
    size_t o = off;
    off = (off + bytes + 255) & ~(size_t)255;
    return o;
  };
  float* hA   = (float*)(base + carve((size_t)kN * kD * 4));
  float* hB   = (float*)(base + carve((size_t)kN * kD * 4));
  float* Ybig = (float*)(base + carve((size_t)kN * 8 * kD * 4));  // Y, later FFN hidden
  float* mbuf = (float*)(base + carve((size_t)kN * kD * 4));      // m, later attn_out
  float* qkv  = (float*)(base + carve((size_t)kN * 192 * 4));
  float* obuf = (float*)(base + carve((size_t)kN * kD * 4));      // o, later ffn2
  float* pin  = (float*)(base + carve((size_t)kN * kD * 4));
  float* pout = (float*)(base + carve((size_t)kN * kD * 4));
  float* comb = (float*)(base + carve((size_t)kG * kCLSK * 4));
  float* c1   = (float*)(base + carve((size_t)kG * kDENSE * 4));
  bf16* wcat  = (bf16*)(base + carve((size_t)kD * 8 * kD * 2));
  bf16* wih   = (bf16*)(base + carve((size_t)kD * 3 * kD * 2));
  bf16* whh   = (bf16*)(base + carve((size_t)kD * 3 * kD * 2));
  bf16* aiw   = (bf16*)(base + carve((size_t)kD * 3 * kD * 2));
  bf16* aow   = (bf16*)(base + carve((size_t)kD * kD * 2));
  bf16* dw1   = (bf16*)(base + carve((size_t)kD * kDENSE * 2));
  bf16* dw2   = (bf16*)(base + carve((size_t)kDENSE * kD * 2));
  bf16* cw1   = (bf16*)(base + carve((size_t)kCLSK * kDENSE * 2));

  const int TB = 256;
  auto blks = [](size_t n, int b) { return (unsigned)((n + b - 1) / b); };

  // --- weight repacks into fragment order ---------------------------------
  pack_B<<<blks(kD * 3 * kD, TB), TB, 0, stream>>>(gru_Wih, wih, 3 * kD, kD, kD);
  pack_B<<<blks(kD * 3 * kD, TB), TB, 0, stream>>>(gru_Whh, whh, 3 * kD, kD, kD);
  pack_B<<<blks(kD * 3 * kD, TB), TB, 0, stream>>>(attn_in_w, aiw, 3 * kD, kD, kD);
  pack_B<<<blks(kD * kD, TB), TB, 0, stream>>>(attn_out_w, aow, kD, kD, kD);
  pack_B<<<blks(kD * kDENSE, TB), TB, 0, stream>>>(dp_W1, dw1, kDENSE, kD, kD);
  pack_B<<<blks(kDENSE * kD, TB), TB, 0, stream>>>(dp_W2, dw2, kD, kDENSE, kDENSE);
  pack_B<<<blks(kCLSK * kDENSE, TB), TB, 0, stream>>>(cls_W1, cw1, kDENSE, kCLSK, kD + kTAB);
  pack_wcat_B<<<blks(kD * 512, TB), TB, 0, stream>>>(W_e, b_e, wcat);

  // --- initial node features ---------------------------------------------
  pad_h0<<<blks((size_t)kN * kD, TB), TB, 0, stream>>>(x, hA);

  const dim3 gblk(32, 4);
  // --- 4 message-passing + GRU steps (ping-pong h) ------------------------
  for (int t = 0; t < 4; ++t) {
    const float* hc = (t & 1) ? hB : hA;
    float*       hn = (t & 1) ? hA : hB;
    // Y[N,512] = h @ Wcat  (rank-8 edge-weight factorization)
    gemm_bf16<<<dim3(kN / 64, 8), gblk, 0, stream>>>(hc, wcat, nullptr, Ybig, 512, kD, 0);
    m_init<<<blks((size_t)kN * 16, TB), TB, 0, stream>>>(mbuf, conv_bias);
    edge_combine<<<blks((size_t)kE * 16, TB), TB, 0, stream>>>(
        Ybig, edge_attr, edge_index, mbuf);
    gru_fused<<<dim3(kN / 16, 4), 32, 0, stream>>>(
        mbuf, hc, hn, wih, whh, gru_bih, gru_bhh);
  }
  // after t=3 the latest h is in hA

  // --- transformer block ---------------------------------------------------
  gemm_bf16<<<dim3(kN / 64, 3), gblk, 0, stream>>>(hA, aiw, attn_in_b, qkv, 192, kD, 0);
  attn_core<<<kG, dim3(kP, 8), 0, stream>>>(qkv, obuf);
  gemm_bf16<<<dim3(kN / 64, 1), gblk, 0, stream>>>(obuf, aow, attn_out_b, mbuf, kD, kD, 0);
  layernorm_res<<<kN / 8, dim3(32, 8), 0, stream>>>(hA, mbuf, ln1_g, ln1_b, pin);
  gemm_bf16<<<dim3(kN / 64, kDENSE / 64), gblk, 0, stream>>>(
      pin, dw1, dp_b1, Ybig, kDENSE, kD, 1);
  gemm_bf16<<<dim3(kN / 64, 1), gblk, 0, stream>>>(Ybig, dw2, dp_b2, obuf, kD, kDENSE, 0);
  layernorm_res<<<kN / 8, dim3(32, 8), 0, stream>>>(pin, obuf, ln2_g, ln2_b, pout);

  // --- readout -------------------------------------------------------------
  pool_concat<<<blks((size_t)kG * kCLSK, TB), TB, 0, stream>>>(pout, u, comb);
  gemm_bf16<<<dim3(kG / 64, kDENSE / 64), gblk, 0, stream>>>(
      comb, cw1, cls_b1, c1, kDENSE, kCLSK, 1);
  final_dot<<<blks(kG, TB), TB, 0, stream>>>(c1, cls_W2, cls_b2, out);
}
